// MultiHeadSelfAttention_63376537419854
// MI455X (gfx1250) — matrix operand order repaired and checked by reading, not compiled, not run on hardware
//
#include <hip/hip_runtime.h>
#include <math.h>

typedef __attribute__((ext_vector_type(16))) _Float16 v16h;
typedef __attribute__((ext_vector_type(8)))  _Float16 v8h;
typedef __attribute__((ext_vector_type(8)))  float    v8f;

constexpr int BB    = 4;
constexpr int CC    = 256;
constexpr int SS    = 2304;     // 48*48
constexpr int HEADS = 8;
constexpr int HD    = 32;
constexpr int STILE = SS / 16;  // 144 token tiles
constexpr int NQW   = 3 * CC * CC;   // qkv_w elements (196608)
constexpr int NPW   = CC * CC;       // proj_w elements (65536)
constexpr float QSCALE = 0.17677669529663689f; // 32^-0.5

// A-fragment (16x32 f16) from a row-major f16 row pointer.
// lane: M = lane%16 (caller bakes into `row`), hh = lane>>4.
// a[j] = row[16*hh + j], a[8+j] = row[16*hh + 8 + j]  (two 16B loads)
__device__ inline v16h load_a_frag(const _Float16* __restrict__ row, int hh) {
  v8h lo = *(const v8h*)(row + 16 * hh);
  v8h hi = *(const v8h*)(row + 16 * hh + 8);
  v16h a;
#pragma unroll
  for (int j = 0; j < 8; ++j) { a[j] = lo[j]; a[8 + j] = hi[j]; }
  return a;
}

// ---------------------------------------------------------------------------
// Prepass A: convert both weight matrices to f16 (one elementwise kernel).
// ---------------------------------------------------------------------------
__global__ void __launch_bounds__(256)
cvt_w_kernel(const float* __restrict__ qkv_w, const float* __restrict__ proj_w,
             _Float16* __restrict__ wq, _Float16* __restrict__ wp) {
  const int i = blockIdx.x * 256 + threadIdx.x;
  if (i < NQW) wq[i] = (_Float16)qkv_w[i];
  else if (i < NQW + NPW) wp[i - NQW] = (_Float16)proj_w[i - NQW];
}

// ---------------------------------------------------------------------------
// Prepass B: x (B,C,S) f32 -> xT (B,S,C) f16 via LDS tile transpose.
// One block per 16(C) x 64(S) tile.
// ---------------------------------------------------------------------------
__global__ void __launch_bounds__(256)
xpose_kernel(const float* __restrict__ x, _Float16* __restrict__ xT) {
  __shared__ _Float16 tile[16][64 + 8];
  const int blk = blockIdx.x;
  const int s64 = blk % (SS / 64);
  const int ct  = (blk / (SS / 64)) % (CC / 16);
  const int b   = blk / ((SS / 64) * (CC / 16));
  const int tid = threadIdx.x;

  const int j = tid & 63;
#pragma unroll
  for (int p = 0; p < 4; ++p) {
    const int i = (tid >> 6) + 4 * p;
    tile[i][j] = (_Float16)x[((size_t)b * CC + ct * 16 + i) * SS + s64 * 64 + j];
  }
  __syncthreads();
  const int i2 = tid & 15;
  const int jg = tid >> 4;
#pragma unroll
  for (int p = 0; p < 4; ++p) {
    const int j2 = jg + 16 * p;
    xT[((size_t)b * SS + s64 * 64 + j2) * CC + ct * 16 + i2] = tile[i2][j2];
  }
}

// ---------------------------------------------------------------------------
// Kernel 1: QKV = xT @ wq^T + qkv_b. One wave per 16(token) x 64(feature)
// tile: 4 accumulators share each A-fragment; 4 v_wmma per K-chunk.
// Q,K stored (B,heads,S,32) f16 (Q pre-scaled); V stored transposed
// (B,heads,32,S) f16 so the flash P@V B-fragments are contiguous.
// ---------------------------------------------------------------------------
__global__ void __launch_bounds__(256)
qkv_gemm_kernel(const _Float16* __restrict__ xT, const _Float16* __restrict__ wq,
                const float* __restrict__ bias,
                _Float16* __restrict__ q, _Float16* __restrict__ k,
                _Float16* __restrict__ vT) {
  const int lane = threadIdx.x & 31;
  const int gw   = blockIdx.x * 8 + (threadIdx.x >> 5);
  const int nt64 = gw % 12;                 // 768/64 feature blocks
  const int st   = (gw / 12) % STILE;       // token tile
  const int b    = gw / (12 * STILE);
  const int n    = lane & 15;
  const int hh   = lane >> 4;

  const _Float16* arow = xT + ((size_t)b * SS + st * 16 + n) * CC;

  v8f acc[4] = {};
  for (int kc = 0; kc < 8; ++kc) {
    const v16h a = load_a_frag(arow + kc * 32, hh);
#pragma unroll
    for (int t = 0; t < 4; ++t) {
      const v16h bf = *(const v16h*)(wq + (size_t)(nt64 * 64 + t * 16 + n) * CC +
                                     kc * 32 + 16 * hh);
      acc[t] = __builtin_amdgcn_wmma_f32_16x16x32_f16(false, a, false, bf,
                                                      (short)0, acc[t], false, false);
    }
  }

#pragma unroll
  for (int t = 0; t < 4; ++t) {
    const int   f    = nt64 * 64 + t * 16 + n;  // output feature in [0,768)
    const float bia  = bias[f];
    const int   which = f >> 8;                 // uniform per t (64|256)
    const int   c    = f & 255;
    const size_t bh  = (size_t)b * HEADS + (c >> 5);
    const int   dim  = c & 31;
    if (which == 0) {
#pragma unroll
      for (int r = 0; r < 8; ++r) {
        const int token = st * 16 + r + 8 * hh;
        q[(bh * SS + token) * HD + dim] = (_Float16)((acc[t][r] + bia) * QSCALE);
      }
    } else if (which == 1) {
#pragma unroll
      for (int r = 0; r < 8; ++r) {
        const int token = st * 16 + r + 8 * hh;
        k[(bh * SS + token) * HD + dim] = (_Float16)(acc[t][r] + bia);
      }
    } else {
#pragma unroll
      for (int r = 0; r < 8; ++r) {
        const int token = st * 16 + r + 8 * hh;
        vT[(bh * HD + dim) * SS + token] = (_Float16)(acc[t][r] + bia);
      }
    }
  }
}

// ---------------------------------------------------------------------------
// Kernel 2: fused flash attention. One wave per (b, head, 16-query block).
// Streams keys 32 at a time; online softmax with 16-lane butterflies; P is
// restaged through per-wave LDS (s_wait_dscnt) into A-fragment layout; V is
// consumed from vT with single contiguous 32B fragment loads.
// ---------------------------------------------------------------------------
__global__ void __launch_bounds__(256)
flash_attn_kernel(const _Float16* __restrict__ q, const _Float16* __restrict__ k,
                  const _Float16* __restrict__ vT, _Float16* __restrict__ attn) {
  __shared__ _Float16 lds[8][16 * 32];
  const int lane = threadIdx.x & 31;
  const int wv   = threadIdx.x >> 5;
  const int gw   = blockIdx.x * 8 + wv;
  const int qt   = gw % STILE;
  const int bh   = gw / STILE;              // b*8 + head
  const int n    = lane & 15;
  const int hh   = lane >> 4;

  const _Float16* kb  = k  + (size_t)bh * SS * HD;
  const _Float16* vt0 = vT + ((size_t)bh * HD + n) * SS;        // dim = n
  const _Float16* vt1 = vT + ((size_t)bh * HD + 16 + n) * SS;   // dim = 16+n

  const v16h qa = load_a_frag(q + ((size_t)bh * SS + qt * 16 + n) * HD, hh);

  v8f o0 = {}, o1 = {};
  float m[8], l[8];
#pragma unroll
  for (int r = 0; r < 8; ++r) { m[r] = -INFINITY; l[r] = 0.0f; }

  _Float16* P = lds[wv];

  for (int kt = 0; kt < STILE; kt += 2) {
    if (kt + 2 < STILE) {                       // prefetch next pair's K/V lines
      const int nb = (kt + 2) * 16;
      __builtin_prefetch(kb + (size_t)(nb + n) * HD, 0, 3);
      __builtin_prefetch(vt0 + nb, 0, 3);
      __builtin_prefetch(vt1 + nb, 0, 3);
    }

    // scores for 32 keys (two 16-wide tiles)
    v8f s0 = {}, s1 = {};
    {
      const v16h bk = *(const v16h*)(kb + (size_t)(kt * 16 + n) * HD + 16 * hh);
      s0 = __builtin_amdgcn_wmma_f32_16x16x32_f16(false, qa, false, bk,
                                                  (short)0, s0, false, false);
    }
    {
      const v16h bk = *(const v16h*)(kb + (size_t)((kt + 1) * 16 + n) * HD + 16 * hh);
      s1 = __builtin_amdgcn_wmma_f32_16x16x32_f16(false, qa, false, bk,
                                                  (short)0, s1, false, false);
    }

    // online softmax per row (row M = r + 8*hh lives across a 16-lane half)
#pragma unroll
    for (int r = 0; r < 8; ++r) {
      float vmax = fmaxf(s0[r], s1[r]);
#pragma unroll
      for (int msk = 8; msk >= 1; msk >>= 1)
        vmax = fmaxf(vmax, __shfl_xor(vmax, msk, 32));
      const float mnew  = fmaxf(m[r], vmax);
      const float alpha = __expf(m[r] - mnew);
      const float p0    = __expf(s0[r] - mnew);
      const float p1    = __expf(s1[r] - mnew);
      float rs = p0 + p1;
#pragma unroll
      for (int msk = 8; msk >= 1; msk >>= 1) rs += __shfl_xor(rs, msk, 32);
      l[r] = l[r] * alpha + rs;
      m[r] = mnew;
      o0[r] *= alpha;
      o1[r] *= alpha;
      P[(r + 8 * hh) * 32 + n]      = (_Float16)p0;
      P[(r + 8 * hh) * 32 + 16 + n] = (_Float16)p1;
    }

    // wave-local LDS ordering: all P stores visible before cross-lane reload
    asm volatile("s_wait_dscnt 0" ::: "memory");

    const v16h pa = load_a_frag(P + n * 32, hh);

    // V B-fragments: single contiguous 32B loads from vT
    const int kbase = kt * 16;
    const v16h bv0 = *(const v16h*)(vt0 + kbase + 16 * hh);
    const v16h bv1 = *(const v16h*)(vt1 + kbase + 16 * hh);
    o0 = __builtin_amdgcn_wmma_f32_16x16x32_f16(false, pa, false, bv0,
                                                (short)0, o0, false, false);
    o1 = __builtin_amdgcn_wmma_f32_16x16x32_f16(false, pa, false, bv1,
                                                (short)0, o1, false, false);
  }

  // normalize and write attn (B, S, C) with C = head*32 + dim
  const int b_   = bh >> 3;
  const int head = bh & 7;
#pragma unroll
  for (int r = 0; r < 8; ++r) {
    const float inv   = 1.0f / l[r];
    const int   token = qt * 16 + r + 8 * hh;
    _Float16* row = attn + ((size_t)b_ * SS + token) * CC + head * HD;
    row[n]      = (_Float16)(o0[r] * inv);
    row[16 + n] = (_Float16)(o1[r] * inv);
  }
}

// ---------------------------------------------------------------------------
// Kernel 3: out = attn @ wp^T + proj_b + x, written (B,C,H,W) f32.
// One wave per 16(token) x 64(feature) tile.
// ---------------------------------------------------------------------------
__global__ void __launch_bounds__(256)
proj_gemm_kernel(const _Float16* __restrict__ attn, const _Float16* __restrict__ wp,
                 const float* __restrict__ bias, const float* __restrict__ x,
                 float* __restrict__ out) {
  const int lane = threadIdx.x & 31;
  const int gw   = blockIdx.x * 8 + (threadIdx.x >> 5);
  const int nt64 = gw % 4;                  // 256/64 feature blocks
  const int st   = (gw / 4) % STILE;
  const int b    = gw / (4 * STILE);
  const int n    = lane & 15;
  const int hh   = lane >> 4;

  const _Float16* arow = attn + ((size_t)b * SS + st * 16 + n) * CC;

  v8f acc[4] = {};
  for (int kc = 0; kc < 8; ++kc) {
    const v16h a = load_a_frag(arow + kc * 32, hh);
#pragma unroll
    for (int t = 0; t < 4; ++t) {
      const v16h bf = *(const v16h*)(wp + (size_t)(nt64 * 64 + t * 16 + n) * CC +
                                     kc * 32 + 16 * hh);
      acc[t] = __builtin_amdgcn_wmma_f32_16x16x32_f16(false, a, false, bf,
                                                      (short)0, acc[t], false, false);
    }
  }

#pragma unroll
  for (int t = 0; t < 4; ++t) {
    const int   cf  = nt64 * 64 + t * 16 + n;
    const float bia = bias[cf];
#pragma unroll
    for (int r = 0; r < 8; ++r) {
      const int    token = st * 16 + r + 8 * hh;
      const size_t idx   = ((size_t)b * CC + cf) * SS + token;
      out[idx] = acc[t][r] + bia + x[idx];
    }
  }
}

// ---------------------------------------------------------------------------
extern "C" void kernel_launch(void* const* d_in, const int* in_sizes, int n_in,
                              void* d_out, int out_size, void* d_ws, size_t ws_size,
                              hipStream_t stream) {
  (void)in_sizes; (void)n_in; (void)out_size; (void)ws_size;
  const float* x      = (const float*)d_in[0];
  const float* qkv_w  = (const float*)d_in[1];
  const float* qkv_b  = (const float*)d_in[2];
  const float* proj_w = (const float*)d_in[3];
  const float* proj_b = (const float*)d_in[4];
  float*       out    = (float*)d_out;

  const size_t nbsc = (size_t)BB * SS * CC;   // 2,359,296 elements
  _Float16* xT   = (_Float16*)d_ws;
  _Float16* wq   = xT + nbsc;
  _Float16* wp   = wq + NQW;
  _Float16* q    = wp + NPW;
  _Float16* k    = q + nbsc;
  _Float16* vT   = k + nbsc;
  _Float16* attn = vT + nbsc;                 // total ~24.1 MB f16

  cvt_w_kernel<<<(NQW + NPW) / 256, 256, 0, stream>>>(qkv_w, proj_w, wq, wp);
  xpose_kernel<<<BB * (CC / 16) * (SS / 64), 256, 0, stream>>>(x, xT);
  qkv_gemm_kernel<<<(BB * STILE * 12) / 8, 256, 0, stream>>>(xT, wq, qkv_b, q, k, vT);
  flash_attn_kernel<<<(BB * HEADS * STILE) / 8, 256, 0, stream>>>(q, k, vT, attn);
  proj_gemm_kernel<<<(BB * STILE * 4) / 8, 256, 0, stream>>>(attn, wp, proj_b, x, out);
}
